// Sage14FX_11708080849318
// MI455X (gfx1250) — matrix operand order repaired and checked by reading, not compiled, not run on hardware
//
#include <hip/hip_runtime.h>
#include <hip/hip_bf16.h>

// ---------------------------------------------------------------------------
// CDNA5 (gfx1250) conv-GRU-attention net.
// All channel contractions -> V_WMMA_F32_16X16X32_F16 (f16 in, f32 acc).
// Weights pre-packed into WMMA B-fragment order (f16, zero-padded) so the
// GEMM hot loops are pure {global/ds}_load_b128 -> v_wmma streams.
// Conv tap staging uses GLOBAL_LOAD_ASYNC_TO_LDS_B128 (ASYNCcnt) so the
// global->LDS copy bypasses VGPRs and overlaps with the WMMA stream.
// Activations stored f16 with K-padded strides (zero-filled) -> no guards.
// ---------------------------------------------------------------------------

typedef __attribute__((ext_vector_type(16))) _Float16 v16h;
typedef __attribute__((ext_vector_type(8)))  _Float16 v8h;
typedef __attribute__((ext_vector_type(8)))  float    v8f;

#define DEVINL __device__ __forceinline__

DEVINL v8f wmma_f16(v16h a, v16h b, v8f c) {
  return __builtin_amdgcn_wmma_f32_16x16x32_f16(false, a, false, b, (short)0, c,
                                                false, false);
}

DEVINL v16h cat8(v8h lo, v8h hi) {
  return __builtin_shufflevector(lo, hi, 0, 1, 2, 3, 4, 5, 6, 7, 8, 9, 10, 11,
                                 12, 13, 14, 15);
}

// Async 16B global->LDS copy (GV mode), tracked by ASYNCcnt.
DEVINL void async_g2l_b128(unsigned lds_off, const void* gaddr) {
  asm volatile("global_load_async_to_lds_b128 %0, %1, off"
               :: "v"(lds_off), "v"((unsigned long long)(uintptr_t)gaddr)
               : "memory");
}
DEVINL void wait_async0() {
  asm volatile("s_wait_asynccnt 0x0" ::: "memory");
}

// Problem constants
#define BB   32
#define TT   8
#define HHs  20
#define WWs  20
#define CIc  10
#define HID  256
#define NIMG (BB * TT)        // 256 encoder images
#define PIX  (HHs * WWs)      // 400 pixels per image
#define CTXD ((TT + 1) * HID) // 2304
#define AD   258
#define ADP  288              // AD padded to 9*32 (K-chunk + tile aligned)
#define GG   20
#define K1P  96               // conv1 K=90 padded to 3*32

// WMMA fragment index maps (ISA 7.12.2, wave32, 16-bit):
// A (16x32): lane L row M=L&15; elem e -> K = (e&7) + ((e>>3)<<4) + ((L>>4)<<3)
//            == two contiguous 8-f16 runs at K = (L>>4)*8 and 16+(L>>4)*8.
// B (32x16): lane L col N=L&15; elem e -> K = (e&7) + ((e>>3)<<4) + ((L>>4)<<3), as A.
// C/D      : lane L col N=L&15; vgpr r -> M = r + ((L>>4)<<3).

// ---------------------------------------------------------------------------
__global__ void k_zero_f32(float* __restrict__ p, int n) {
  int i = blockIdx.x * blockDim.x + threadIdx.x;
  if (i < n) p[i] = 0.f;
}

// ---------------------------------------------------------------------------
// Repack one f32 weight matrix W[K,N] (row major) into f16 WMMA B-fragments:
// Bp[((t*NT + j)*32 + lane)*16 + e] = W[t*32 + (e&7) + ((e>>3)<<4) + ((lane>>4)<<3)][j*16+(lane&15)]
// Out-of-range (K,N) -> 0, so consumers need no guards. grid (KT,NT), block 32.
__global__ void k_packw(const float* __restrict__ W, int K, int N, int NT,
                        _Float16* __restrict__ Bp) {
  int t = blockIdx.x, j = blockIdx.y, lane = threadIdx.x & 31;
  int n = j * 16 + (lane & 15);
  int kb = t * 32 + ((lane >> 4) << 3);
  v16h v;
#pragma unroll
  for (int e = 0; e < 16; ++e) {
    int k = kb + (e & 7) + ((e >> 3) << 4);
    v[e] = (k < K && n < N) ? (_Float16)W[(size_t)k * N + n] : (_Float16)0.f;
  }
  *(v16h*)(Bp + ((size_t)(t * NT + j) * 32 + lane) * 16) = v;
}

// ---------------------------------------------------------------------------
// conv1 im2col: x[NIMG,20,20,10] f32 -> col[M=NIMG*PIX, 96] f16 (90 + 6 zeros)
__global__ void k_im2col1(const float* __restrict__ x, _Float16* __restrict__ col) {
  int row = blockIdx.x, e = threadIdx.x;  // block 96
  int img = row / PIX, p = row % PIX;
  int py = p / WWs, px = p % WWs;
  float v = 0.f;
  if (e < 90) {
    int tap = e / CIc, ci = e % CIc;
    int iy = py + tap / 3 - 1, ix = px + tap % 3 - 1;
    if (iy >= 0 && iy < HHs && ix >= 0 && ix < WWs)
      v = x[(((size_t)img * HHs + iy) * WWs + ix) * CIc + ci];
  }
  col[(size_t)row * K1P + e] = (_Float16)v;
}

// ---------------------------------------------------------------------------
// Generic packed-weight WMMA GEMM.
// A: f16 [M, lda], lda multiple of 32, rows fully zero-padded beyond Keff.
// Bp: packed fragments (KT x NT tiles). Writes NT*16 columns (pad cols get 0).
// Optional bias/relu; optional column pooling into sumbuf[(m/grp)*N + n].
__global__ void k_gemm_p(const _Float16* __restrict__ A, int lda,
                         const _Float16* __restrict__ Bp, int KT, int NT,
                         const float* __restrict__ bias, int N,
                         _Float16* __restrict__ out, int ldo, int relu,
                         float* __restrict__ sumbuf, int grp) {
  int mt = blockIdx.x, nt = blockIdx.y;
  int lane = threadIdx.x & 31, half = lane >> 4, l15 = lane & 15;
  const _Float16* arow = A + (size_t)(mt * 16 + l15) * lda + (half << 3);
  const _Float16* bbase = Bp + ((size_t)nt * 32 + lane) * 16;
  v8f acc = {};
  for (int t = 0; t < KT; ++t) {
    v8h alo = *(const v8h*)(arow + t * 32);
    v8h ahi = *(const v8h*)(arow + t * 32 + 16);
    v16h b = *(const v16h*)(bbase + (size_t)t * NT * 32 * 16);
    acc = wmma_f16(cat8(alo, ahi), b, acc);
  }
  int n = nt * 16 + l15;
  float bs = (bias && n < N) ? bias[n] : 0.f;
#pragma unroll
  for (int r = 0; r < 8; ++r) {
    int m = mt * 16 + r + (half << 3);
    float v = acc[r] + bs;
    if (relu && v < 0.f) v = 0.f;
    if (n >= N) v = 0.f;  // keep K-pad columns zero for downstream GEMMs
    out[(size_t)m * ldo + n] = (_Float16)v;
    if (sumbuf && n < N) atomicAdd(&sumbuf[(size_t)(m / grp) * N + n], v);
  }
}

// ---------------------------------------------------------------------------
// Encoder conv2 (3x3, 256->256) + relu + per-pixel LayerNorm + pooled
// accumulation into feats_sum. 16 waves/block own the 16 n-tiles of one
// 16-pixel M tile. Tap-major: im2col rows staged in LDS once per tap via
// async global->LDS b128 (halo lanes zero-filled with ds_store), then
// ds_load_b128 fragment reads give 16x A-reuse across waves.
__global__ void __launch_bounds__(512)
k_conv2ln(const _Float16* __restrict__ act1, const _Float16* __restrict__ Bp,
          const float* __restrict__ bias, const float* __restrict__ g,
          const float* __restrict__ bb, float* __restrict__ feats_sum) {
  __shared__ __align__(16) _Float16 As[16][HID];
  __shared__ float tile[16][HID];
  __shared__ float mu[16], rs[16];
  int mt = blockIdx.x;
  int lane = threadIdx.x & 31;
  int half = lane >> 4, l15 = lane & 15;
  int nt = threadIdx.x >> 5;
  // staging assignment: thread owns 8 f16 of row (tid>>5) at col (tid&31)*8
  int sr = threadIdx.x >> 5;
  int sc = (threadIdx.x & 31) * 8;
  int srow = mt * 16 + sr;
  int simg = srow / PIX, sp = srow % PIX;
  int spy = sp / WWs, spx = sp % WWs;
  unsigned lds_off = (unsigned)(uintptr_t)&As[sr][sc];
  const _Float16* bbase = Bp + ((size_t)nt * 32 + lane) * 16;
  v8f acc = {};
  for (int tap = 0; tap < 9; ++tap) {
    int iy = spy + tap / 3 - 1, ix = spx + tap % 3 - 1;
    bool valid = (iy >= 0 && iy < HHs && ix >= 0 && ix < WWs);
    __syncthreads();  // previous tap's fragment reads done
    if (valid) {
      async_g2l_b128(lds_off,
                     act1 + (((size_t)simg * HHs + iy) * WWs + ix) * HID + sc);
    } else {
      v8h z = {};
      *(v8h*)(&As[sr][sc]) = z;
    }
    wait_async0();
    __syncthreads();
    const _Float16* arow = &As[l15][half << 3];
#pragma unroll
    for (int c0 = 0; c0 < HID; c0 += 32) {
      v8h alo = *(const v8h*)(arow + c0);
      v8h ahi = *(const v8h*)(arow + c0 + 16);
      int t = tap * 8 + (c0 >> 5);
      v16h b = *(const v16h*)(bbase + (size_t)t * 16 * 32 * 16);  // NT=16
      acc = wmma_f16(cat8(alo, ahi), b, acc);
    }
  }
  int n = nt * 16 + l15;
#pragma unroll
  for (int r = 0; r < 8; ++r) {
    float v = acc[r] + bias[n];
    tile[r + (half << 3)][n] = v > 0.f ? v : 0.f;
  }
  __syncthreads();
  if (threadIdx.x < 16) {
    int q = threadIdx.x;
    float s = 0.f;
    for (int c = 0; c < HID; ++c) s += tile[q][c];
    float m = s / HID;
    float vv = 0.f;
    for (int c = 0; c < HID; ++c) { float d = tile[q][c] - m; vv += d * d; }
    mu[q] = m; rs[q] = rsqrtf(vv / HID + 1e-3f);
  }
  __syncthreads();
  for (int e = threadIdx.x; e < 16 * HID; e += blockDim.x) {
    int q = e >> 8, c = e & 255;
    int im = (mt * 16 + q) / PIX;
    float v = (tile[q][c] - mu[q]) * rs[q] * g[c] + bb[c];
    atomicAdd(&feats_sum[(size_t)im * HID + c], v);
  }
}

// ---------------------------------------------------------------------------
// GRU (keras reset_after), one block per batch element; writes f16 ctx.
__global__ void k_gru(const float* __restrict__ feats_sum,
                      const float* __restrict__ wk, const float* __restrict__ wr,
                      const float* __restrict__ bi, const float* __restrict__ br,
                      _Float16* __restrict__ ctx) {
  __shared__ float h[HID], xt[HID];
  int b = blockIdx.x, c = threadIdx.x;
  h[c] = 0.f;
  float hn = 0.f;
  for (int t = 0; t < TT; ++t) {
    xt[c] = feats_sum[((size_t)(b * TT + t)) * HID + c] * (1.f / PIX);
    __syncthreads();
    float xz = bi[c], xr = bi[HID + c], xh = bi[2 * HID + c];
    float hz = br[c], hr = br[HID + c], hh = br[2 * HID + c];
    for (int k = 0; k < HID; ++k) {
      float xv = xt[k], hv = h[k];
      const float* wkk = wk + (size_t)k * 3 * HID;
      const float* wrk = wr + (size_t)k * 3 * HID;
      xz += xv * wkk[c];           hz += hv * wrk[c];
      xr += xv * wkk[HID + c];     hr += hv * wrk[HID + c];
      xh += xv * wkk[2 * HID + c]; hh += hv * wrk[2 * HID + c];
    }
    float z  = 1.f / (1.f + expf(-(xz + hz)));
    float r  = 1.f / (1.f + expf(-(xr + hr)));
    float cg = tanhf(xh + r * hh);
    hn = z * h[c] + (1.f - z) * cg;
    __syncthreads();
    h[c] = hn;
    ctx[(size_t)b * CTXD + HID + t * HID + c] = (_Float16)hn;
  }
  ctx[(size_t)b * CTXD + c] = (_Float16)hn;
}

// ---------------------------------------------------------------------------
__global__ void k_pos(const float* __restrict__ pw, const float* __restrict__ pb,
                      _Float16* __restrict__ pe) {
  int p = blockIdx.x, c = threadIdx.x;
  int py = p / GG, px = p % GG;
  float yy = -1.f + 2.f * py / (GG - 1);
  float xx = -1.f + 2.f * px / (GG - 1);
  pe[(size_t)p * HID + c] = (_Float16)tanhf(yy * pw[c] + xx * pw[HID + c] + pb[c]);
}

// ---------------------------------------------------------------------------
// Attention: logits via q=qc[b]+qp[p], k=kc[b]+kp[p]; softmax over 400 pixels;
// writes attv[b*400+p][0..287] f16 (cols >=258 zeroed for the next GEMM).
__global__ void __launch_bounds__(512)
k_attn(const _Float16* __restrict__ qc, const _Float16* __restrict__ kc,
       const _Float16* __restrict__ vc, const _Float16* __restrict__ qp,
       const _Float16* __restrict__ kp, const _Float16* __restrict__ vp,
       _Float16* __restrict__ attv) {
  __shared__ float lgt[PIX];
  __shared__ float red[2];
  int b = blockIdx.x, t = threadIdx.x;
  for (int p = t; p < PIX; p += blockDim.x) {
    float s = 0.f;
    for (int c = 0; c < AD; ++c) {
      float qv = (float)qc[(size_t)b * ADP + c] + (float)qp[(size_t)p * ADP + c];
      float kv = (float)kc[(size_t)b * ADP + c] + (float)kp[(size_t)p * ADP + c];
      s += qv * kv;
    }
    lgt[p] = s;
  }
  __syncthreads();
  if (t == 0) {
    float mx = -3.4e38f;
    for (int p = 0; p < PIX; ++p) mx = fmaxf(mx, lgt[p]);
    float sm = 0.f;
    for (int p = 0; p < PIX; ++p) sm += expf(lgt[p] - mx);
    red[0] = mx; red[1] = sm;
  }
  __syncthreads();
  float mx = red[0], inv = 1.f / red[1];
  for (int p = t; p < PIX; p += blockDim.x) {
    float a = expf(lgt[p] - mx) * inv;
    size_t base = ((size_t)b * PIX + p) * ADP;
    for (int c = 0; c < ADP; ++c) {
      float v = (c < AD)
                    ? ((float)vc[(size_t)b * ADP + c] + (float)vp[(size_t)p * ADP + c]) * a
                    : 0.f;
      attv[base + c] = (_Float16)v;
    }
  }
}

// ---------------------------------------------------------------------------
__global__ void k_sel(const float* __restrict__ xpsum, const float* __restrict__ sw,
                      const float* __restrict__ sb, float* __restrict__ wsel) {
  __shared__ float lg[BB][4];
  int t = threadIdx.x;
  if (t < BB * 4) {
    int b = t >> 2, j = t & 3;
    float s = sb[j];
    for (int c = 0; c < HID; ++c)
      s += (xpsum[(size_t)b * HID + c] * (1.f / PIX)) * sw[c * 4 + j];
    lg[b][j] = s;
  }
  __syncthreads();
  if (t < BB) {
    float mx = lg[t][0];
    for (int j = 1; j < 4; ++j) mx = fmaxf(mx, lg[t][j]);
    float e0[4], sm = 0.f;
    for (int j = 0; j < 4; ++j) { e0[j] = expf(lg[t][j] - mx); sm += e0[j]; }
    for (int j = 0; j < 4; ++j) wsel[t * 4 + j] = e0[j] / sm;
  }
}

// ---------------------------------------------------------------------------
// 4-hypothesis mixture: A fragments loaded once per K-chunk, 4 packed-B
// streams + 4 WMMA accumulators, combined with wsel (tile shares batch index).
__global__ void k_cand(const _Float16* __restrict__ xp, const _Float16* __restrict__ Bp,
                       const float* __restrict__ whb, const float* __restrict__ wsel,
                       _Float16* __restrict__ chosen) {
  int mt = blockIdx.x, nt = blockIdx.y;
  int lane = threadIdx.x & 31, half = lane >> 4, l15 = lane & 15;
  const _Float16* arow = xp + (size_t)(mt * 16 + l15) * HID + (half << 3);
  const _Float16* bbase = Bp + ((size_t)nt * 32 + lane) * 16;
  const size_t jstride = (size_t)8 * 16 * 32 * 16;  // KT=8, NT=16 per hypothesis
  v8f acc[4] = {};
  for (int t = 0; t < 8; ++t) {
    v16h a = cat8(*(const v8h*)(arow + t * 32), *(const v8h*)(arow + t * 32 + 16));
#pragma unroll
    for (int j = 0; j < 4; ++j) {
      v16h b = *(const v16h*)(bbase + jstride * j + (size_t)t * 16 * 32 * 16);
      acc[j] = wmma_f16(a, b, acc[j]);
    }
  }
  int n = nt * 16 + l15;
  int img = (mt * 16) / PIX;
#pragma unroll
  for (int r = 0; r < 8; ++r) {
    int m = mt * 16 + r + (half << 3);
    float v = 0.f;
#pragma unroll
    for (int j = 0; j < 4; ++j) {
      float c = acc[j][r] + whb[j * HID + n];
      if (c < 0.f) c = 0.f;
      v += c * wsel[img * 4 + j];
    }
    chosen[(size_t)m * HID + n] = (_Float16)v;
  }
}

// ---------------------------------------------------------------------------
// Decoder: 3x3 conv (packed weights, async-LDS-staged taps) + relu + LN +
// 256->10 dense, fused. Same block structure as k_conv2ln.
__global__ void __launch_bounds__(512)
k_dec(const _Float16* __restrict__ chosen, const _Float16* __restrict__ Bp,
      const float* __restrict__ b1, const float* __restrict__ lg,
      const float* __restrict__ lb, const float* __restrict__ w2,
      const float* __restrict__ b2, float* __restrict__ out) {
  __shared__ __align__(16) _Float16 As[16][HID];
  __shared__ float tile[16][HID];
  __shared__ float mu[16], rs[16];
  int mt = blockIdx.x;
  int lane = threadIdx.x & 31;
  int half = lane >> 4, l15 = lane & 15;
  int nt = threadIdx.x >> 5;
  int sr = threadIdx.x >> 5;
  int sc = (threadIdx.x & 31) * 8;
  int srow = mt * 16 + sr;
  int simg = srow / PIX, sp = srow % PIX;
  int spy = sp / WWs, spx = sp % WWs;
  unsigned lds_off = (unsigned)(uintptr_t)&As[sr][sc];
  const _Float16* bbase = Bp + ((size_t)nt * 32 + lane) * 16;
  v8f acc = {};
  for (int tap = 0; tap < 9; ++tap) {
    int iy = spy + tap / 3 - 1, ix = spx + tap % 3 - 1;
    bool valid = (iy >= 0 && iy < HHs && ix >= 0 && ix < WWs);
    __syncthreads();
    if (valid) {
      async_g2l_b128(lds_off,
                     chosen + (((size_t)simg * HHs + iy) * WWs + ix) * HID + sc);
    } else {
      v8h z = {};
      *(v8h*)(&As[sr][sc]) = z;
    }
    wait_async0();
    __syncthreads();
    const _Float16* arow = &As[l15][half << 3];
#pragma unroll
    for (int c0 = 0; c0 < HID; c0 += 32) {
      v8h alo = *(const v8h*)(arow + c0);
      v8h ahi = *(const v8h*)(arow + c0 + 16);
      int t = tap * 8 + (c0 >> 5);
      v16h b = *(const v16h*)(bbase + (size_t)t * 16 * 32 * 16);
      acc = wmma_f16(cat8(alo, ahi), b, acc);
    }
  }
  int n = nt * 16 + l15;
#pragma unroll
  for (int r = 0; r < 8; ++r) {
    float v = acc[r] + b1[n];
    tile[r + (half << 3)][n] = v > 0.f ? v : 0.f;
  }
  __syncthreads();
  if (threadIdx.x < 16) {
    int q = threadIdx.x;
    float s = 0.f;
    for (int c = 0; c < HID; ++c) s += tile[q][c];
    float m = s / HID;
    float vv = 0.f;
    for (int c = 0; c < HID; ++c) { float d = tile[q][c] - m; vv += d * d; }
    mu[q] = m; rs[q] = rsqrtf(vv / HID + 1e-3f);
  }
  __syncthreads();
  for (int e = threadIdx.x; e < 16 * HID; e += blockDim.x) {
    int q = e >> 8, c = e & 255;
    tile[q][c] = (tile[q][c] - mu[q]) * rs[q] * lg[c] + lb[c];
  }
  __syncthreads();
  for (int t = threadIdx.x; t < 16 * 10; t += blockDim.x) {
    int q = t / 10, j = t % 10;
    float s = b2[j];
    for (int c = 0; c < HID; ++c) s += tile[q][c] * w2[c * 10 + j];
    out[(size_t)(mt * 16 + q) * 10 + j] = s;
  }
}

// ---------------------------------------------------------------------------
extern "C" void kernel_launch(void* const* d_in, const int* in_sizes, int n_in,
                              void* d_out, int out_size, void* d_ws, size_t ws_size,
                              hipStream_t stream) {
  (void)in_sizes; (void)n_in; (void)out_size; (void)ws_size;
  const float* x_seq    = (const float*)d_in[0];
  const float* enc_w1   = (const float*)d_in[1];
  const float* enc_b1   = (const float*)d_in[2];
  const float* enc_w2   = (const float*)d_in[3];
  const float* enc_b2   = (const float*)d_in[4];
  const float* ln_g     = (const float*)d_in[5];
  const float* ln_b     = (const float*)d_in[6];
  const float* gru_wk   = (const float*)d_in[7];
  const float* gru_wr   = (const float*)d_in[8];
  const float* gru_bi   = (const float*)d_in[9];
  const float* gru_br   = (const float*)d_in[10];
  const float* pos_w    = (const float*)d_in[11];
  const float* pos_b    = (const float*)d_in[12];
  const float* q_w      = (const float*)d_in[13];
  const float* q_b      = (const float*)d_in[14];
  const float* k_w      = (const float*)d_in[15];
  const float* k_b      = (const float*)d_in[16];
  const float* v_w      = (const float*)d_in[17];
  const float* v_b      = (const float*)d_in[18];
  const float* ao_w     = (const float*)d_in[19];
  const float* ao_b     = (const float*)d_in[20];
  const float* ch_in_w  = (const float*)d_in[21];
  const float* ch_in_b  = (const float*)d_in[22];
  const float* ch_h_w   = (const float*)d_in[23];
  const float* ch_h_b   = (const float*)d_in[24];
  const float* sel_w    = (const float*)d_in[25];
  const float* sel_b    = (const float*)d_in[26];
  const float* dec_w1   = (const float*)d_in[27];
  const float* dec_b1   = (const float*)d_in[28];
  const float* dec_ln_g = (const float*)d_in[29];
  const float* dec_ln_b = (const float*)d_in[30];
  const float* dec_w2   = (const float*)d_in[31];
  const float* dec_b2   = (const float*)d_in[32];

  char* ws = (char*)d_ws;
  size_t off = 0;
  auto alloc = [&](size_t bytes) -> void* {
    void* p = ws + off;
    off = (off + bytes + 255) & ~(size_t)255;
    return p;
  };
  const int M = NIMG * PIX;    // 102400 encoder rows
  const int MD = BB * PIX;     // 12800 decoder rows
  // activations
  _Float16* col1    = (_Float16*)alloc((size_t)M * K1P * 2);
  _Float16* act1    = (_Float16*)alloc((size_t)M * HID * 2);
  float*    featsum = (float*)alloc((size_t)NIMG * HID * 4);
  _Float16* ctx     = (_Float16*)alloc((size_t)BB * CTXD * 2);
  _Float16* pe      = (_Float16*)alloc((size_t)PIX * HID * 2);
  _Float16* qc      = (_Float16*)alloc((size_t)BB * ADP * 2);
  _Float16* kc      = (_Float16*)alloc((size_t)BB * ADP * 2);
  _Float16* vc      = (_Float16*)alloc((size_t)BB * ADP * 2);
  _Float16* qp      = (_Float16*)alloc((size_t)PIX * ADP * 2);
  _Float16* kp      = (_Float16*)alloc((size_t)PIX * ADP * 2);
  _Float16* vp      = (_Float16*)alloc((size_t)PIX * ADP * 2);
  _Float16* attv    = (_Float16*)alloc((size_t)MD * ADP * 2);
  _Float16* attb    = (_Float16*)alloc((size_t)MD * ADP * 2);
  _Float16* xp      = (_Float16*)alloc((size_t)MD * HID * 2);
  float*    xpsum   = (float*)alloc((size_t)BB * HID * 4);
  float*    wsel    = (float*)alloc((size_t)BB * 16);
  _Float16* chosen  = (_Float16*)alloc((size_t)MD * HID * 2);
  // packed weights (fragments: KT*NT*32*16 f16)
  auto pk = [&](int KT, int NT) { return (_Float16*)alloc((size_t)KT * NT * 512 * 2); };
  _Float16* P_w1   = pk(3, 16);            // enc conv1, K=96
  _Float16* P_w2   = pk(72, 16);           // enc conv2, K=2304
  _Float16* P_qc   = pk(72, 17);           // q_w ctx part
  _Float16* P_kc   = pk(72, 17);
  _Float16* P_vc   = pk(72, 17);
  _Float16* P_qp   = pk(8, 17);            // q_w pe part
  _Float16* P_kp   = pk(8, 17);
  _Float16* P_vp   = pk(8, 17);
  _Float16* P_ao   = pk(9, 18);            // NT=18: zero-pads cols 258..287
  _Float16* P_ci   = pk(9, 16);
  _Float16* P_ch   = pk(8 * 4, 16);        // 4 hypotheses back-to-back
  _Float16* P_d1   = pk(72, 16);           // dec conv1

  // --- weight packing (f32 -> f16 B-fragment order) ---
  const size_t peoff = (size_t)CTXD * AD;
  k_packw<<<dim3(3, 16), 32, 0, stream>>>(enc_w1, 9 * CIc, HID, 16, P_w1);
  k_packw<<<dim3(72, 16), 32, 0, stream>>>(enc_w2, 9 * HID, HID, 16, P_w2);
  k_packw<<<dim3(72, 17), 32, 0, stream>>>(q_w, CTXD, AD, 17, P_qc);
  k_packw<<<dim3(72, 17), 32, 0, stream>>>(k_w, CTXD, AD, 17, P_kc);
  k_packw<<<dim3(72, 17), 32, 0, stream>>>(v_w, CTXD, AD, 17, P_vc);
  k_packw<<<dim3(8, 17), 32, 0, stream>>>(q_w + peoff, HID, AD, 17, P_qp);
  k_packw<<<dim3(8, 17), 32, 0, stream>>>(k_w + peoff, HID, AD, 17, P_kp);
  k_packw<<<dim3(8, 17), 32, 0, stream>>>(v_w + peoff, HID, AD, 17, P_vp);
  k_packw<<<dim3(9, 18), 32, 0, stream>>>(ao_w, AD, AD, 18, P_ao);
  k_packw<<<dim3(9, 16), 32, 0, stream>>>(ch_in_w, AD, HID, 16, P_ci);
  for (int j = 0; j < 4; ++j)
    k_packw<<<dim3(8, 16), 32, 0, stream>>>(ch_h_w + (size_t)j * HID * HID, HID,
                                            HID, 16, P_ch + (size_t)j * 8 * 16 * 512);
  k_packw<<<dim3(72, 16), 32, 0, stream>>>(dec_w1, 9 * HID, HID, 16, P_d1);

  // --- zero accumulators every call (graph-replayed) ---
  k_zero_f32<<<(NIMG * HID + 255) / 256, 256, 0, stream>>>(featsum, NIMG * HID);
  k_zero_f32<<<(BB * HID + 255) / 256, 256, 0, stream>>>(xpsum, BB * HID);

  // --- encoder ---
  k_im2col1<<<M, K1P, 0, stream>>>(x_seq, col1);
  k_gemm_p<<<dim3(M / 16, 16), 32, 0, stream>>>(col1, K1P, P_w1, 3, 16, enc_b1,
                                                HID, act1, HID, 1, nullptr, 1);
  k_conv2ln<<<M / 16, 512, 0, stream>>>(act1, P_w2, enc_b2, ln_g, ln_b, featsum);

  // --- GRU + positional ---
  k_gru<<<BB, HID, 0, stream>>>(featsum, gru_wk, gru_wr, gru_bi, gru_br, ctx);
  k_pos<<<PIX, HID, 0, stream>>>(pos_w, pos_b, pe);

  // --- q/k/v projections (ctx part [32x2304] + pe part [400x256]) ---
  dim3 gcx(BB / 16, 17), gpe(PIX / 16, 17);
  k_gemm_p<<<gcx, 32, 0, stream>>>(ctx, CTXD, P_qc, 72, 17, q_b, AD, qc, ADP, 0, nullptr, 1);
  k_gemm_p<<<gcx, 32, 0, stream>>>(ctx, CTXD, P_kc, 72, 17, k_b, AD, kc, ADP, 0, nullptr, 1);
  k_gemm_p<<<gcx, 32, 0, stream>>>(ctx, CTXD, P_vc, 72, 17, v_b, AD, vc, ADP, 0, nullptr, 1);
  k_gemm_p<<<gpe, 32, 0, stream>>>(pe, HID, P_qp, 8, 17, nullptr, AD, qp, ADP, 0, nullptr, 1);
  k_gemm_p<<<gpe, 32, 0, stream>>>(pe, HID, P_kp, 8, 17, nullptr, AD, kp, ADP, 0, nullptr, 1);
  k_gemm_p<<<gpe, 32, 0, stream>>>(pe, HID, P_vp, 8, 17, nullptr, AD, vp, ADP, 0, nullptr, 1);

  // --- attention softmax + a*v ---
  k_attn<<<BB, 512, 0, stream>>>(qc, kc, vc, qp, kp, vp, attv);

  // --- ao_w projection; ch_in + relu + pooling ---
  k_gemm_p<<<dim3(MD / 16, 18), 32, 0, stream>>>(attv, ADP, P_ao, 9, 18, ao_b,
                                                 AD, attb, ADP, 0, nullptr, 1);
  k_gemm_p<<<dim3(MD / 16, 16), 32, 0, stream>>>(attb, ADP, P_ci, 9, 16, ch_in_b,
                                                 HID, xp, HID, 1, xpsum, PIX);

  // --- hypothesis selection + mixture ---
  k_sel<<<1, 128, 0, stream>>>(xpsum, sel_w, sel_b, wsel);
  k_cand<<<dim3(MD / 16, 16), 32, 0, stream>>>(xp, P_ch, ch_h_b, wsel, chosen);

  // --- decoder conv + LN + final dense ---
  k_dec<<<MD / 16, 512, 0, stream>>>(chosen, P_d1, dec_b1, dec_ln_g, dec_ln_b,
                                     dec_w2, dec_b2, (float*)d_out);
}
